// GPSA_62466004353685
// MI455X (gfx1250) — matrix operand order repaired and checked by reading, not compiled, not run on hardware
//
#include <hip/hip_runtime.h>
#include <stdint.h>

typedef __attribute__((ext_vector_type(16))) __bf16 v16bf;
typedef __attribute__((ext_vector_type(8)))  float  v8f;

constexpr int Bc = 16, Nc = 576, Dc = 1024, Hc = 16, HDc = 64;
constexpr int Mc = Bc * Nc;            // 9216 rows
constexpr int NT64 = Nc / 64;          // 9 row-tiles per (b,h) in attention

__device__ __forceinline__ unsigned short f2bf(float f) {
  union { float f; unsigned u; } v; v.f = f;
  unsigned r = v.u + 0x7FFFu + ((v.u >> 16) & 1u);
  return (unsigned short)(r >> 16);
}

// ---------------- fp32 -> bf16 convert (8 elems / thread, b128 in/out) ----------
__global__ void cvt_bf16(const float* __restrict__ in, unsigned short* __restrict__ out, int n) {
  int i = (blockIdx.x * 256 + threadIdx.x) * 8;
  if (i >= n) return;
  float4 a = *(const float4*)(in + i);
  float4 b = *(const float4*)(in + i + 4);
  union { unsigned short s[8]; uint4 u; } o;
  o.s[0] = f2bf(a.x); o.s[1] = f2bf(a.y); o.s[2] = f2bf(a.z); o.s[3] = f2bf(a.w);
  o.s[4] = f2bf(b.x); o.s[5] = f2bf(b.y); o.s[6] = f2bf(b.z); o.s[7] = f2bf(b.w);
  *(uint4*)(out + i) = o.u;
}

// ---------------- positional softmax: [H,N,N] bf16, computed once ----------------
__global__ void pos_softmax(const float* __restrict__ rel, const float* __restrict__ Wp,
                            const float* __restrict__ bp, unsigned short* __restrict__ pos) {
  __shared__ float red[256];
  int bid = blockIdx.x;
  int h = bid / Nc, n = bid % Nc;
  float w0 = Wp[0 * Hc + h], w1 = Wp[1 * Hc + h], w2 = Wp[2 * Hc + h], bb = bp[h];
  float lo[3];
  float mx = -1e30f;
  #pragma unroll
  for (int t = 0; t < 3; ++t) {
    int m = threadIdx.x + t * 256;
    float l = -1e30f;
    if (m < Nc) {
      const float* r = rel + ((size_t)n * Nc + m) * 3;
      l = r[0] * w0 + r[1] * w1 + r[2] * w2 + bb;
    }
    lo[t] = l; mx = fmaxf(mx, l);
  }
  red[threadIdx.x] = mx; __syncthreads();
  for (int s = 128; s > 0; s >>= 1) {
    if (threadIdx.x < s) red[threadIdx.x] = fmaxf(red[threadIdx.x], red[threadIdx.x + s]);
    __syncthreads();
  }
  mx = red[0]; __syncthreads();
  float sm = 0.f;
  #pragma unroll
  for (int t = 0; t < 3; ++t) {
    int m = threadIdx.x + t * 256;
    if (m < Nc) { lo[t] = __expf(lo[t] - mx); sm += lo[t]; }
  }
  red[threadIdx.x] = sm; __syncthreads();
  for (int s = 128; s > 0; s >>= 1) {
    if (threadIdx.x < s) red[threadIdx.x] += red[threadIdx.x + s];
    __syncthreads();
  }
  float inv = 1.f / red[0];
  unsigned short* prow = pos + ((size_t)h * Nc + n) * Nc;
  #pragma unroll
  for (int t = 0; t < 3; ++t) {
    int m = threadIdx.x + t * 256;
    if (m < Nc) prow[m] = f2bf(lo[t] * inv);
  }
}

// ---------------- bf16 WMMA GEMM: C[128x64] per block, 8 waves, K-step 64 ----------
// MODE 1: x@W_qk -> q/k split per-head layout [B,H,N,HD]
// MODE 2: x@W_v  -> v transposed [B,H,HD,N]
// MODE 3: attn_out@W_proj + bias -> fp32 out [M,Nout]
template <int MODE>
__global__ __launch_bounds__(256) void gemm_bf16(
    const unsigned short* __restrict__ A, const unsigned short* __restrict__ Bw,
    int K, int Nout,
    unsigned short* __restrict__ o_q, unsigned short* __restrict__ o_k,
    unsigned short* __restrict__ o_vt,
    float* __restrict__ o_f, const float* __restrict__ bias) {
  __shared__ unsigned short At[128 * 72];   // pitch 72 shorts (144B) keeps 16B alignment
  __shared__ unsigned short Bt[64 * 72];    // transposed B tile: [n][k]
  const int tid = threadIdx.x;
  const int n0 = blockIdx.x * 64;
  const int r0 = blockIdx.y * 128;
  const int wv = tid >> 5, lane = tid & 31;
  const int wm = wv & 3, wn = wv >> 2;      // 4 (M) x 2 (N) waves
  const int half = lane >> 4, l16 = lane & 15;

  v8f z = {};
  v8f acc[2][2];
  #pragma unroll
  for (int i = 0; i < 2; ++i)
    #pragma unroll
    for (int j = 0; j < 2; ++j) acc[i][j] = z;

  const int arow = tid >> 1, akc = (tid & 1) * 32;
  const int bkr  = tid >> 2, bnc = (tid & 3) * 16;

  union AF { uint4 u[2]; v16bf v; };

  for (int kb = 0; kb < K; kb += 64) {
    { // A tile 128x64: 64B per thread
      const unsigned short* src = A + (size_t)(r0 + arow) * K + kb + akc;
      *(uint4*)(&At[arow * 72 + akc])      = *(const uint4*)(src);
      *(uint4*)(&At[arow * 72 + akc + 8])  = *(const uint4*)(src + 8);
      *(uint4*)(&At[arow * 72 + akc + 16]) = *(const uint4*)(src + 16);
      *(uint4*)(&At[arow * 72 + akc + 24]) = *(const uint4*)(src + 24);
      if (kb + 64 < K) __builtin_prefetch(src + 64, 0, 3);
    }
    { // B tile 64x64, stored transposed
      const unsigned short* src = Bw + (size_t)(kb + bkr) * Nout + n0 + bnc;
      uint4 p0 = *(const uint4*)(src);
      uint4 p1 = *(const uint4*)(src + 8);
      const unsigned short* e0 = (const unsigned short*)&p0;
      const unsigned short* e1 = (const unsigned short*)&p1;
      #pragma unroll
      for (int i = 0; i < 8; ++i) Bt[(bnc + i) * 72 + bkr]     = e0[i];
      #pragma unroll
      for (int i = 0; i < 8; ++i) Bt[(bnc + 8 + i) * 72 + bkr] = e1[i];
      if (kb + 64 < K) __builtin_prefetch(src + (size_t)64 * Nout, 0, 3);
    }
    __syncthreads();
    AF af[2][2], bf[2][2];
    #pragma unroll
    for (int mt = 0; mt < 2; ++mt) {
      int rl = wm * 32 + mt * 16 + l16;
      #pragma unroll
      for (int kbi = 0; kbi < 2; ++kbi) {
        af[mt][kbi].u[0] = *(const uint4*)(&At[rl * 72 + kbi * 32 + half * 8]);
        af[mt][kbi].u[1] = *(const uint4*)(&At[rl * 72 + kbi * 32 + 16 + half * 8]);
      }
    }
    #pragma unroll
    for (int nt = 0; nt < 2; ++nt) {
      int nc = wn * 32 + nt * 16 + l16;
      #pragma unroll
      for (int kbi = 0; kbi < 2; ++kbi) {
        bf[nt][kbi].u[0] = *(const uint4*)(&Bt[nc * 72 + kbi * 32 + half * 8]);
        bf[nt][kbi].u[1] = *(const uint4*)(&Bt[nc * 72 + kbi * 32 + 16 + half * 8]);
      }
    }
    #pragma unroll
    for (int kbi = 0; kbi < 2; ++kbi)
      #pragma unroll
      for (int mt = 0; mt < 2; ++mt)
        #pragma unroll
        for (int nt = 0; nt < 2; ++nt)
          acc[mt][nt] = __builtin_amdgcn_wmma_f32_16x16x32_bf16(
              false, af[mt][kbi].v, false, bf[nt][kbi].v, (short)0, acc[mt][nt], false, false);
    __syncthreads();
  }

  #pragma unroll
  for (int mt = 0; mt < 2; ++mt) {
    const int rowb = r0 + wm * 32 + mt * 16 + half * 8;   // 8-row run never crosses batch
    const int bidx = rowb / Nc;
    const int nbase = rowb % Nc;
    #pragma unroll
    for (int nt = 0; nt < 2; ++nt) {
      const int col = n0 + wn * 32 + nt * 16 + l16;
      #pragma unroll
      for (int j = 0; j < 8; ++j) {
        float v = acc[mt][nt][j];
        if (MODE == 1) {
          int cc = col & (Dc - 1);
          int h = cc >> 6, hd = cc & 63;
          unsigned short* dst = (col < Dc) ? o_q : o_k;
          dst[(((size_t)bidx * Hc + h) * Nc + nbase + j) * HDc + hd] = f2bf(v);
        } else if (MODE == 2) {
          int h = col >> 6, hd = col & 63;
          o_vt[(((size_t)bidx * Hc + h) * HDc + hd) * Nc + nbase + j] = f2bf(v);
        } else {
          o_f[(size_t)(rowb + j) * Nout + col] = v + bias[col];
        }
      }
    }
  }
}

// ---------------- fused GPSA attention ----------------
// Block = 4 waves; wave owns 16 query rows. Online softmax over 64-col tiles.
// Row max via lane-half butterflies; row SUM via WMMA against all-ones B
// (accS rescales with the same factor as the output accumulator).
__global__ __launch_bounds__(128) void gpsa_attn(
    const unsigned short* __restrict__ q, const unsigned short* __restrict__ k,
    const unsigned short* __restrict__ vt, const unsigned short* __restrict__ pos,
    const float* __restrict__ gating, unsigned short* __restrict__ aout) {
  __shared__ unsigned short Pls[4][16 * 72];   // per-wave P-tile scratch (16x64, pitch 72)
  const int tid = threadIdx.x;
  const int wv = tid >> 5, lane = tid & 31;
  const int half = lane >> 4, l16 = lane & 15;
  const int bid = blockIdx.x;
  const int mt0 = bid % NT64;
  const int h = (bid / NT64) % Hc;
  const int b = bid / (NT64 * Hc);
  const int m0 = mt0 * 64 + wv * 16;
  const size_t bh = (size_t)b * Hc + h;
  const unsigned short* qb  = q  + bh * Nc * HDc;
  const unsigned short* kbp = k  + bh * Nc * HDc;
  const unsigned short* vtb = vt + bh * HDc * Nc;
  const unsigned short* posb = pos + (size_t)h * Nc * Nc;

  union AF { uint4 u[2]; v16bf v; };

  AF qf[2];                                    // q rows, two K-steps over HD=64
  #pragma unroll
  for (int kbi = 0; kbi < 2; ++kbi) {
    const unsigned short* src = qb + (size_t)(m0 + l16) * HDc + kbi * 32 + half * 8;
    qf[kbi].u[0] = *(const uint4*)(src);
    qf[kbi].u[1] = *(const uint4*)(src + 16);
  }

  AF onesf;                                    // bf16 1.0 everywhere
  {
    uint4 o1; o1.x = o1.y = o1.z = o1.w = 0x3F803F80u;
    onesf.u[0] = o1; onesf.u[1] = o1;
  }

  v8f z = {};
  v8f accP[4], accG[4], accS = z;
  #pragma unroll
  for (int i = 0; i < 4; ++i) { accP[i] = z; accG[i] = z; }
  float mrun[8];
  #pragma unroll
  for (int j = 0; j < 8; ++j) mrun[j] = -1e30f;

  unsigned short* Pw = Pls[wv];

  for (int nj = 0; nj < Nc; nj += 64) {
    // S tile (16x64) = q @ k^T
    v8f s[4];
    #pragma unroll
    for (int nt = 0; nt < 4; ++nt) {
      v8f a = z;
      #pragma unroll
      for (int kbi = 0; kbi < 2; ++kbi) {
        AF bk;
        const unsigned short* src = kbp + (size_t)(nj + nt * 16 + l16) * HDc + kbi * 32 + half * 8;
        bk.u[0] = *(const uint4*)(src);
        bk.u[1] = *(const uint4*)(src + 16);
        a = __builtin_amdgcn_wmma_f32_16x16x32_bf16(false, qf[kbi].v, false, bk.v,
                                                    (short)0, a, false, false);
      }
      s[nt] = a;
    }
    // online max + rescale; row-sum deferred to WMMA-with-ones
    #pragma unroll
    for (int j = 0; j < 8; ++j) {
      float v0 = s[0][j] * 0.125f, v1 = s[1][j] * 0.125f;   // scale = HD^-0.5
      float v2 = s[2][j] * 0.125f, v3 = s[3][j] * 0.125f;
      float t = fmaxf(fmaxf(v0, v1), fmaxf(v2, v3));
      #pragma unroll
      for (int msk = 8; msk > 0; msk >>= 1) t = fmaxf(t, __shfl_xor(t, msk, 16));
      float mnew = fmaxf(mrun[j], t);
      float r = __expf(mrun[j] - mnew);
      mrun[j] = mnew;
      accS[j] *= r;
      #pragma unroll
      for (int nt2 = 0; nt2 < 4; ++nt2) accP[nt2][j] *= r;
      int row = half * 8 + j;
      Pw[row * 72 + l16]      = f2bf(__expf(v0 - mnew));
      Pw[row * 72 + 16 + l16] = f2bf(__expf(v1 - mnew));
      Pw[row * 72 + 32 + l16] = f2bf(__expf(v2 - mnew));
      Pw[row * 72 + 48 + l16] = f2bf(__expf(v3 - mnew));
    }
    asm volatile("s_wait_dscnt 0" ::: "memory");   // wave-local LDS RAW fence
    AF pa[2], ga[2];
    #pragma unroll
    for (int kbi = 0; kbi < 2; ++kbi) {
      pa[kbi].u[0] = *(const uint4*)(&Pw[l16 * 72 + kbi * 32 + half * 8]);
      pa[kbi].u[1] = *(const uint4*)(&Pw[l16 * 72 + kbi * 32 + 16 + half * 8]);
      const unsigned short* src = posb + (size_t)(m0 + l16) * Nc + nj + kbi * 32 + half * 8;
      ga[kbi].u[0] = *(const uint4*)(src);
      ga[kbi].u[1] = *(const uint4*)(src + 16);
    }
    // softmax denominator: accS += P @ ones
    accS = __builtin_amdgcn_wmma_f32_16x16x32_bf16(false, pa[0].v, false, onesf.v,
                                                   (short)0, accS, false, false);
    accS = __builtin_amdgcn_wmma_f32_16x16x32_bf16(false, pa[1].v, false, onesf.v,
                                                   (short)0, accS, false, false);
    #pragma unroll
    for (int nt2 = 0; nt2 < 4; ++nt2) {
      #pragma unroll
      for (int kbi = 0; kbi < 2; ++kbi) {
        AF vb;
        const unsigned short* src = vtb + (size_t)(nt2 * 16 + l16) * Nc + nj + kbi * 32 + half * 8;
        vb.u[0] = *(const uint4*)(src);
        vb.u[1] = *(const uint4*)(src + 16);
        accP[nt2] = __builtin_amdgcn_wmma_f32_16x16x32_bf16(false, pa[kbi].v, false, vb.v,
                                                            (short)0, accP[nt2], false, false);
        accG[nt2] = __builtin_amdgcn_wmma_f32_16x16x32_bf16(false, ga[kbi].v, false, vb.v,
                                                            (short)0, accG[nt2], false, false);
      }
    }
  }
  float g = 1.f / (1.f + __expf(-gating[h]));
  float a_[8];
  #pragma unroll
  for (int j = 0; j < 8; ++j) a_[j] = (1.f - g) / accS[j];
  #pragma unroll
  for (int nt2 = 0; nt2 < 4; ++nt2) {
    #pragma unroll
    for (int j = 0; j < 8; ++j) {
      int row = m0 + half * 8 + j;
      int dcol = h * HDc + nt2 * 16 + l16;
      float v = a_[j] * accP[nt2][j] + g * accG[nt2][j];
      aout[(size_t)(b * Nc + row) * Dc + dcol] = f2bf(v);
    }
  }
}

// ---------------- launch ----------------
extern "C" void kernel_launch(void* const* d_in, const int* in_sizes, int n_in,
                              void* d_out, int out_size, void* d_ws, size_t ws_size,
                              hipStream_t stream) {
  const float* x      = (const float*)d_in[0];
  const float* W_qk   = (const float*)d_in[1];
  const float* W_v    = (const float*)d_in[2];
  const float* W_proj = (const float*)d_in[3];
  const float* b_proj = (const float*)d_in[4];
  const float* W_pos  = (const float*)d_in[5];
  const float* b_pos  = (const float*)d_in[6];
  const float* gating = (const float*)d_in[7];
  const float* rel    = (const float*)d_in[8];
  float* out = (float*)d_out;

  char* ws = (char*)d_ws;
  size_t off = 0;
  auto alloc = [&](size_t elems) -> unsigned short* {
    unsigned short* p = (unsigned short*)(ws + off);
    off += elems * 2;
    return p;
  };
  unsigned short* xb   = alloc((size_t)Mc * Dc);
  unsigned short* wqkb = alloc((size_t)Dc * 2 * Dc);
  unsigned short* wvb  = alloc((size_t)Dc * Dc);
  unsigned short* wpb  = alloc((size_t)Dc * Dc);
  unsigned short* qb   = alloc((size_t)Mc * Dc);
  unsigned short* kb   = alloc((size_t)Mc * Dc);
  unsigned short* vtb  = alloc((size_t)Mc * Dc);
  unsigned short* posb = alloc((size_t)Hc * Nc * Nc);
  unsigned short* aob  = alloc((size_t)Mc * Dc);

  int nx = Mc * Dc, nqk = Dc * 2 * Dc, nv = Dc * Dc;
  cvt_bf16<<<nx  / (256 * 8), 256, 0, stream>>>(x,      xb,   nx);
  cvt_bf16<<<nqk / (256 * 8), 256, 0, stream>>>(W_qk,   wqkb, nqk);
  cvt_bf16<<<nv  / (256 * 8), 256, 0, stream>>>(W_v,    wvb,  nv);
  cvt_bf16<<<nv  / (256 * 8), 256, 0, stream>>>(W_proj, wpb,  nv);

  pos_softmax<<<Hc * Nc, 256, 0, stream>>>(rel, W_pos, b_pos, posb);

  gemm_bf16<1><<<dim3(2 * Dc / 64, Mc / 128), 256, 0, stream>>>(
      xb, wqkb, Dc, 2 * Dc, qb, kb, nullptr, nullptr, nullptr);
  gemm_bf16<2><<<dim3(Dc / 64, Mc / 128), 256, 0, stream>>>(
      xb, wvb, Dc, Dc, nullptr, nullptr, vtb, nullptr, nullptr);

  gpsa_attn<<<Bc * Hc * NT64, 128, 0, stream>>>(qb, kb, vtb, posb, gating, aob);

  gemm_bf16<3><<<dim3(Dc / 64, Mc / 128), 256, 0, stream>>>(
      aob, wpb, Dc, Dc, nullptr, nullptr, nullptr, out, b_proj);
}
